// Repro_7189775254155
// MI455X (gfx1250) — matrix operand order repaired and checked by reading, not compiled, not run on hardware
//
#include <hip/hip_runtime.h>
#include <hip/hip_bf16.h>
#include <stdint.h>

#define SEQ   512
#define EMB   128
#define HID   4096
#define HEADS 64
#define HDIM  64
#define FFN   16384

typedef __attribute__((ext_vector_type(16))) __bf16       v16bf;
typedef __attribute__((ext_vector_type(8)))  float        v8f;
typedef __attribute__((ext_vector_type(4)))  unsigned int v4u;
typedef __attribute__((ext_vector_type(2)))  unsigned int v2u;
typedef __attribute__((ext_vector_type(4)))  float        v4f;
typedef __attribute__((ext_vector_type(2)))  float        v2f;
typedef __attribute__((ext_vector_type(4)))  int          v4i;

#if __has_builtin(__builtin_amdgcn_global_load_async_to_lds_b128)
#define USE_ASYNC_LDS 1
// pointer-to-AS1 v4i and pointer-to-AS3 v4i (per the builtin's signature)
typedef __attribute__((address_space(1))) v4i* gv4i_p;
typedef __attribute__((address_space(3))) v4i* lv4i_p;
#else
#define USE_ASYNC_LDS 0
#endif

__device__ __forceinline__ void wait_async0() {
#if __has_builtin(__builtin_amdgcn_s_wait_asynccnt)
  __builtin_amdgcn_s_wait_asynccnt(0);
#else
  asm volatile("s_wait_asynccnt 0x0" ::: "memory");
#endif
}

// ---- f32 -> bf16 (round-to-nearest, cheap): add 0x8000 then take hi16 ----
__device__ __forceinline__ unsigned short f2bf_s(float f) {
  unsigned u = __builtin_bit_cast(unsigned, f) + 0x8000u;
  return (unsigned short)(u >> 16);
}
#if __has_builtin(__builtin_amdgcn_perm)
__device__ __forceinline__ unsigned pk2(float a, float b) {
  unsigned ua = __builtin_bit_cast(unsigned, a) + 0x8000u;
  unsigned ub = __builtin_bit_cast(unsigned, b) + 0x8000u;
  // D = [ua.b2, ua.b3, ub.b2, ub.b3]  (one v_perm_b32)
  return __builtin_amdgcn_perm(ub, ua, 0x07060302u);
}
#else
__device__ __forceinline__ unsigned pk2(float a, float b) {
  return (unsigned)f2bf_s(a) | ((unsigned)f2bf_s(b) << 16);
}
#endif

__device__ __forceinline__ float gelu_tanh_f(float x) {
  return 0.5f * x * (1.0f + tanhf(0.7978845608028654f * (x + 0.044715f * x * x * x)));
}

#define BM 256
#define BN 64
#define BK 32
#define GFLAG_GELU 1

// ---------------------------------------------------------------------------
// WMMA GEMM: C[z] = A[z] (MxK bf16, row-major, lda) * W[z]^T + bias
//   W (f32): element (n,k) at Bw + z*sBz + n*sBn + k*sBk (sBk==1 -> vec path)
// Epilogue: optional GELU, optional f32 residual add, stores f32 C (Cf) and/or
// bf16 C (Cb16). Grid: (M/BM, N/BN, batch).
// A tile: GLOBAL_LOAD_ASYNC_TO_LDS_B128 (bf16, no conversion, ASYNCcnt).
// B tile: register-staged f32 -> bf16 (v_perm pack) -> ds_store, behind WMMAs.
// 8 waves x (32x64) output strips, double-buffered LDS.
// ---------------------------------------------------------------------------
__global__ __launch_bounds__(256) void wmma_gemm(
    const unsigned short* __restrict__ A, int lda, long long sAz,
    const float* __restrict__ Bw, long long sBn, long long sBk, long long sBz,
    const float* __restrict__ bias,
    const float* __restrict__ resid,
    float* __restrict__ Cf, unsigned short* __restrict__ Cb16,
    int ldc, long long sCz,
    int K, int flags)
{
  __shared__ __align__(16) unsigned int lA[2][BM][20];  // 32 bf16 + pad per row
  __shared__ __align__(16) unsigned int lB[2][BN][20];

  const int tid = threadIdx.x;
  const int z   = blockIdx.z;
  const long long rowBase = (long long)blockIdx.x * BM;
  const long long colBase = (long long)blockIdx.y * BN;
  const unsigned short* Ab = A + sAz * z;
  const float*          Bb = Bw + sBz * z;

  const int arow = tid >> 1, acol = (tid & 1) * 16;   // A fallback: 2 thr x 2 rows
  const int brow = tid >> 2, bcol = (tid & 3) * 8;    // B: 4 threads per row
  const bool bvec = (sBk == 1);

#if !USE_ASYNC_LDS
  v4u  ga[2][2];
#endif
  v4f  gb[2];
  float gbs[8];

  // ---- A tile: async memory->LDS copy, one thread per row, 4x b128 ----
  auto stageA = [&](int buf, int k0) {
#if USE_ASYNC_LDS
    const unsigned short* ap = Ab + (rowBase + tid) * (long long)lda + k0;
    gv4i_p g = (gv4i_p)(uintptr_t)ap;
    lv4i_p l = (lv4i_p)(unsigned int)(uintptr_t)&lA[buf][tid][0];
    __builtin_amdgcn_global_load_async_to_lds_b128(g, l,  0, 0);
    __builtin_amdgcn_global_load_async_to_lds_b128(g, l, 16, 0);
    __builtin_amdgcn_global_load_async_to_lds_b128(g, l, 32, 0);
    __builtin_amdgcn_global_load_async_to_lds_b128(g, l, 48, 0);
#else
#pragma unroll
    for (int rr = 0; rr < 2; ++rr) {
      const unsigned short* ap =
          Ab + (rowBase + arow + rr * 128) * (long long)lda + k0 + acol;
      ga[rr][0] = *(const v4u*)(ap);
      ga[rr][1] = *(const v4u*)(ap + 8);
    }
#endif
  };
#if !USE_ASYNC_LDS
  auto commitA = [&](int buf) {
#pragma unroll
    for (int rr = 0; rr < 2; ++rr) {
      *(v4u*)&lA[buf][arow + rr * 128][(tid & 1) * 8 + 0] = ga[rr][0];
      *(v4u*)&lA[buf][arow + rr * 128][(tid & 1) * 8 + 4] = ga[rr][1];
    }
  };
#endif

  auto fetchB = [&](int k0) {
    if (bvec) {
      const float* bp = Bb + (colBase + brow) * sBn + k0 + bcol;
      gb[0] = *(const v4f*)bp;
      gb[1] = *(const v4f*)(bp + 4);
    } else {
      const float* bp = Bb + (colBase + brow) * sBn + (long long)(k0 + bcol) * sBk;
#pragma unroll
      for (int j = 0; j < 8; ++j) gbs[j] = bp[(long long)j * sBk];
    }
  };
  auto commitB = [&](int buf) {
    if (bvec) {
#pragma unroll
      for (int i = 0; i < 2; ++i) {
        v2u p; p.x = pk2(gb[i].x, gb[i].y); p.y = pk2(gb[i].z, gb[i].w);
        *(v2u*)&lB[buf][brow][(tid & 3) * 4 + 2 * i] = p;
      }
    } else {
      v2u p0; p0.x = pk2(gbs[0], gbs[1]); p0.y = pk2(gbs[2], gbs[3]);
      v2u p1; p1.x = pk2(gbs[4], gbs[5]); p1.y = pk2(gbs[6], gbs[7]);
      *(v2u*)&lB[buf][brow][(tid & 3) * 4 + 0] = p0;
      *(v2u*)&lB[buf][brow][(tid & 3) * 4 + 2] = p1;
    }
  };

  const int w    = tid >> 5;      // wave id 0..7 -> 32-row M sub-strip
  const int lane = tid & 31;
  const int m    = lane & 15;     // row (A) / col (B) within 16x16 tile
  const int hf   = lane >> 4;     // K-half selector (contiguous k = 16*hf + s)

  v8f acc[2][4] = {};

  const int nk = K / BK;
  stageA(0, 0);
  fetchB(0);
  commitB(0);
#if !USE_ASYNC_LDS
  commitA(0);
#endif
  for (int ks = 0; ks < nk; ++ks) {
#if USE_ASYNC_LDS
    wait_async0();                    // our async LDS writes visible pre-barrier
#endif
    __syncthreads();
    const int cur  = ks & 1;
    const bool more = (ks + 1 < nk);
    if (more) {
      stageA(cur ^ 1, (ks + 1) * BK); // async A copy + B loads fly during WMMAs
      fetchB((ks + 1) * BK);
    }

    union F { v4u u[2]; v16bf v; };
    F a0, a1, b0, b1, b2, b3;
    a0.u[0] = *(const v4u*)&lA[cur][w * 32 + m][hf * 8];
    a0.u[1] = *(const v4u*)&lA[cur][w * 32 + m][hf * 8 + 4];
    a1.u[0] = *(const v4u*)&lA[cur][w * 32 + 16 + m][hf * 8];
    a1.u[1] = *(const v4u*)&lA[cur][w * 32 + 16 + m][hf * 8 + 4];
    b0.u[0] = *(const v4u*)&lB[cur][ 0 + m][hf * 8];
    b0.u[1] = *(const v4u*)&lB[cur][ 0 + m][hf * 8 + 4];
    b1.u[0] = *(const v4u*)&lB[cur][16 + m][hf * 8];
    b1.u[1] = *(const v4u*)&lB[cur][16 + m][hf * 8 + 4];
    b2.u[0] = *(const v4u*)&lB[cur][32 + m][hf * 8];
    b2.u[1] = *(const v4u*)&lB[cur][32 + m][hf * 8 + 4];
    b3.u[0] = *(const v4u*)&lB[cur][48 + m][hf * 8];
    b3.u[1] = *(const v4u*)&lB[cur][48 + m][hf * 8 + 4];

    acc[0][0] = __builtin_amdgcn_wmma_f32_16x16x32_bf16(false, a0.v, false, b0.v, (short)0, acc[0][0], false, false);
    acc[0][1] = __builtin_amdgcn_wmma_f32_16x16x32_bf16(false, a0.v, false, b1.v, (short)0, acc[0][1], false, false);
    acc[0][2] = __builtin_amdgcn_wmma_f32_16x16x32_bf16(false, a0.v, false, b2.v, (short)0, acc[0][2], false, false);
    acc[0][3] = __builtin_amdgcn_wmma_f32_16x16x32_bf16(false, a0.v, false, b3.v, (short)0, acc[0][3], false, false);
    acc[1][0] = __builtin_amdgcn_wmma_f32_16x16x32_bf16(false, a1.v, false, b0.v, (short)0, acc[1][0], false, false);
    acc[1][1] = __builtin_amdgcn_wmma_f32_16x16x32_bf16(false, a1.v, false, b1.v, (short)0, acc[1][1], false, false);
    acc[1][2] = __builtin_amdgcn_wmma_f32_16x16x32_bf16(false, a1.v, false, b2.v, (short)0, acc[1][2], false, false);
    acc[1][3] = __builtin_amdgcn_wmma_f32_16x16x32_bf16(false, a1.v, false, b3.v, (short)0, acc[1][3], false, false);

    if (more) {
      commitB(cur ^ 1);               // B convert + ds_store behind the WMMAs
#if !USE_ASYNC_LDS
      commitA(cur ^ 1);
#endif
    }
  }

  // Epilogue: C VGPR r -> (M = hf*8 + r, N = m) within each 16x16 tile
#pragma unroll
  for (int mi = 0; mi < 2; ++mi) {
#pragma unroll
    for (int nt = 0; nt < 4; ++nt) {
      const long long gn = colBase + nt * 16 + m;
      const float bvl = bias ? bias[gn] : 0.0f;
#pragma unroll
      for (int r = 0; r < 8; ++r) {
        const long long gm = rowBase + w * 32 + mi * 16 + hf * 8 + r;
        float val = acc[mi][nt][r] + bvl;
        if (flags & GFLAG_GELU) val = gelu_tanh_f(val);
        const long long idx = (gm * (long long)ldc + gn) + sCz * z;
        if (resid) val += resid[idx];
        if (Cf)    Cf[idx] = val;
        if (Cb16)  Cb16[idx] = f2bf_s(val);
      }
    }
  }
}

// ---------------------------------------------------------------------------
// Embedding gather + LayerNorm over EMB=128 -> bf16. One block per token.
// ---------------------------------------------------------------------------
__global__ __launch_bounds__(128) void embed_ln(
    const float* __restrict__ we, const float* __restrict__ te,
    const float* __restrict__ pe, const float* __restrict__ g,
    const float* __restrict__ b, const int* __restrict__ ids,
    const int* __restrict__ tts, const int* __restrict__ pos,
    unsigned short* __restrict__ out)
{
  const int s = blockIdx.x, t = threadIdx.x;
  const int id = ids[s], tt = tts[s], ps = pos[s];
  float v = we[(long long)id * EMB + t] + te[(long long)tt * EMB + t]
          + pe[(long long)ps * EMB + t];
  __shared__ float red[128];
  red[t] = v; __syncthreads();
  for (int st = 64; st > 0; st >>= 1) { if (t < st) red[t] += red[t + st]; __syncthreads(); }
  const float mean = red[0] * (1.0f / EMB); __syncthreads();
  const float d = v - mean;
  red[t] = d * d; __syncthreads();
  for (int st = 64; st > 0; st >>= 1) { if (t < st) red[t] += red[t + st]; __syncthreads(); }
  const float inv = rsqrtf(red[0] * (1.0f / EMB) + 1e-12f);
  out[(long long)s * EMB + t] = f2bf_s(d * inv * g[t] + b[t]);
}

// ---------------------------------------------------------------------------
// Row softmax over 512 columns: f32 in (raw scores), bf16 out (probs).
// Applies 1/8 scale; additive mask is identically 0. Thread t owns cols 2t,2t+1.
// ---------------------------------------------------------------------------
__global__ __launch_bounds__(256) void softmax_rows_bf(
    const float* __restrict__ x, unsigned short* __restrict__ o, float scale)
{
  const float* p = x + (long long)blockIdx.x * SEQ;
  unsigned* po = (unsigned*)(o + (long long)blockIdx.x * SEQ);
  const int t = threadIdx.x;
  const v2f vv = *(const v2f*)(p + 2 * t);
  const float v0 = vv.x * scale, v1 = vv.y * scale;
  __shared__ float red[256];
  red[t] = fmaxf(v0, v1); __syncthreads();
  for (int st = 128; st > 0; st >>= 1) { if (t < st) red[t] = fmaxf(red[t], red[t + st]); __syncthreads(); }
  const float mx = red[0]; __syncthreads();
  const float e0 = __expf(v0 - mx), e1 = __expf(v1 - mx);
  red[t] = e0 + e1; __syncthreads();
  for (int st = 128; st > 0; st >>= 1) { if (t < st) red[t] += red[t + st]; __syncthreads(); }
  const float inv = 1.0f / red[0];
  po[t] = pk2(e0 * inv, e1 * inv);
}

// ---------------------------------------------------------------------------
// LayerNorm over width=4096 -> f32 and/or bf16. One block (256 thr) per row.
// ---------------------------------------------------------------------------
__global__ __launch_bounds__(256) void layernorm_rows(
    const float* __restrict__ x, const float* __restrict__ g,
    const float* __restrict__ b, float* __restrict__ out,
    unsigned short* __restrict__ out_bf, int width)
{
  const int row = blockIdx.x, t = threadIdx.x;
  const float* p = x + (long long)row * width;
  float vals[16];
  float s = 0.0f;
#pragma unroll
  for (int i = 0; i < 16; ++i) { vals[i] = p[t + i * 256]; s += vals[i]; }
  __shared__ float red[256];
  red[t] = s; __syncthreads();
  for (int st = 128; st > 0; st >>= 1) { if (t < st) red[t] += red[t + st]; __syncthreads(); }
  const float mean = red[0] / (float)width; __syncthreads();
  float vs = 0.0f;
#pragma unroll
  for (int i = 0; i < 16; ++i) { float d = vals[i] - mean; vs += d * d; }
  red[t] = vs; __syncthreads();
  for (int st = 128; st > 0; st >>= 1) { if (t < st) red[t] += red[t + st]; __syncthreads(); }
  const float inv = rsqrtf(red[0] / (float)width + 1e-12f);
#pragma unroll
  for (int i = 0; i < 16; ++i) {
    const int c = t + i * 256;
    const float o = (vals[i] - mean) * inv * g[c] + b[c];
    if (out)    out[(long long)row * width + c] = o;
    if (out_bf) out_bf[(long long)row * width + c] = f2bf_s(o);
  }
}

// ---------------------------------------------------------------------------
extern "C" void kernel_launch(void* const* d_in, const int* in_sizes, int n_in,
                              void* d_out, int out_size, void* d_ws, size_t ws_size,
                              hipStream_t stream)
{
  const float* word_emb = (const float*)d_in[0];
  const float* tok_emb  = (const float*)d_in[1];
  const float* pos_emb  = (const float*)d_in[2];
  const float* ln0_g    = (const float*)d_in[3];
  const float* ln0_b    = (const float*)d_in[4];
  const float* W_proj   = (const float*)d_in[5];
  const float* b_proj   = (const float*)d_in[6];
  const float* Wq       = (const float*)d_in[7];
  const float* bq       = (const float*)d_in[8];
  const float* Wk       = (const float*)d_in[9];
  const float* bk       = (const float*)d_in[10];
  const float* Wv       = (const float*)d_in[11];
  const float* bv       = (const float*)d_in[12];
  const float* Wo       = (const float*)d_in[13];
  const float* bo       = (const float*)d_in[14];
  const float* ln1_g    = (const float*)d_in[15];
  const float* ln1_b    = (const float*)d_in[16];
  const float* W_ff1    = (const float*)d_in[17];
  const float* b_ff1    = (const float*)d_in[18];
  const float* W_ff2    = (const float*)d_in[19];
  const float* b_ff2    = (const float*)d_in[20];
  const float* ln2_g    = (const float*)d_in[21];
  const float* ln2_b    = (const float*)d_in[22];
  const int*   in_ids   = (const int*)d_in[23];
  const int*   tt_ids   = (const int*)d_in[24];
  const int*   ps_ids   = (const int*)d_in[25];
  float* out = (float*)d_out;

  // ---- workspace carve (bump allocator, 256B aligned) ----
  char* base = (char*)d_ws;
  size_t off = 0;
  auto alloc = [&](size_t bytes) -> char* {
    char* p = base + off;
    off += (bytes + 255) & ~(size_t)255;
    return p;
  };
  float*          hidden    = (float*)alloc((size_t)SEQ * HID * 4);
  float*          kbuf      = (float*)alloc((size_t)SEQ * HID * 4);
  float*          vbuf      = (float*)alloc((size_t)SEQ * HID * 4);
  float*          scores    = (float*)alloc((size_t)HEADS * SEQ * SEQ * 4);
  float*          s1        = (float*)alloc((size_t)SEQ * HID * 4);
  float*          h1        = (float*)alloc((size_t)SEQ * HID * 4);
  float*          s2        = (float*)alloc((size_t)SEQ * HID * 4);
  unsigned short* h_e_bf    = (unsigned short*)alloc((size_t)SEQ * EMB * 2);
  unsigned short* hidden_bf = (unsigned short*)alloc((size_t)SEQ * HID * 2);
  unsigned short* q_bf      = (unsigned short*)alloc((size_t)SEQ * HID * 2);
  unsigned short* probs_bf  = (unsigned short*)alloc((size_t)HEADS * SEQ * SEQ * 2);
  unsigned short* ctx_bf    = (unsigned short*)alloc((size_t)SEQ * HID * 2);
  unsigned short* h1_bf     = (unsigned short*)alloc((size_t)SEQ * HID * 2);
  unsigned short* ffact_bf  = (unsigned short*)alloc((size_t)SEQ * FFN * 2);

  const dim3 blk(256);

  // 1) embeddings + LN0 -> bf16
  embed_ln<<<SEQ, 128, 0, stream>>>(word_emb, tok_emb, pos_emb, ln0_g, ln0_b,
                                    in_ids, tt_ids, ps_ids, h_e_bf);

  // 2) hidden = h_e @ W_proj^T + b_proj      [512,4096], K=128 (f32 + bf16 out)
  wmma_gemm<<<dim3(SEQ / BM, HID / BN, 1), blk, 0, stream>>>(
      h_e_bf, EMB, 0, W_proj, EMB, 1, 0, b_proj, nullptr,
      hidden, hidden_bf, HID, 0, EMB, 0);

  // 3) q/k/v = hidden @ W^T + b              [512,4096], K=4096
  wmma_gemm<<<dim3(SEQ / BM, HID / BN, 1), blk, 0, stream>>>(
      hidden_bf, HID, 0, Wq, HID, 1, 0, bq, nullptr,
      nullptr, q_bf, HID, 0, HID, 0);
  wmma_gemm<<<dim3(SEQ / BM, HID / BN, 1), blk, 0, stream>>>(
      hidden_bf, HID, 0, Wk, HID, 1, 0, bk, nullptr,
      kbuf, nullptr, HID, 0, HID, 0);
  wmma_gemm<<<dim3(SEQ / BM, HID / BN, 1), blk, 0, stream>>>(
      hidden_bf, HID, 0, Wv, HID, 1, 0, bv, nullptr,
      vbuf, nullptr, HID, 0, HID, 0);

  // 4) scores[h] = q[h] @ k[h]^T             per head: [512,512], K=64
  wmma_gemm<<<dim3(SEQ / BM, SEQ / BN, HEADS), blk, 0, stream>>>(
      q_bf, HID, HDIM, kbuf, HID, 1, HDIM, nullptr, nullptr,
      scores, nullptr, SEQ, (long long)SEQ * SEQ, HDIM, 0);

  // 5) probs = softmax(scores/8) -> bf16
  softmax_rows_bf<<<HEADS * SEQ, blk, 0, stream>>>(scores, probs_bf, 0.125f);

  // 6) ctx[h] = probs[h] @ v[h]              per head: [512,64], K=512 (B strided)
  wmma_gemm<<<dim3(SEQ / BM, HDIM / BN, HEADS), blk, 0, stream>>>(
      probs_bf, SEQ, (long long)SEQ * SEQ, vbuf, 1, HID, HDIM, nullptr, nullptr,
      nullptr, ctx_bf, HID, HDIM, SEQ, 0);

  // 7) s1 = ctx @ Wo^T + bo + hidden         [512,4096], K=4096
  wmma_gemm<<<dim3(SEQ / BM, HID / BN, 1), blk, 0, stream>>>(
      ctx_bf, HID, 0, Wo, HID, 1, 0, bo, hidden,
      s1, nullptr, HID, 0, HID, 0);

  // 8) h1 = LN1(s1) -> f32 + bf16
  layernorm_rows<<<SEQ, blk, 0, stream>>>(s1, ln1_g, ln1_b, h1, h1_bf, HID);

  // 9) ffact = gelu(h1 @ W_ff1^T + b_ff1)    [512,16384], K=4096 (bf16 only)
  wmma_gemm<<<dim3(SEQ / BM, FFN / BN, 1), blk, 0, stream>>>(
      h1_bf, HID, 0, W_ff1, HID, 1, 0, b_ff1, nullptr,
      nullptr, ffact_bf, FFN, 0, HID, GFLAG_GELU);

  // 10) s2 = ffact @ W_ff2^T + b_ff2 + h1    [512,4096], K=16384
  wmma_gemm<<<dim3(SEQ / BM, HID / BN, 1), blk, 0, stream>>>(
      ffact_bf, FFN, 0, W_ff2, FFN, 1, 0, b_ff2, h1,
      s2, nullptr, HID, 0, FFN, 0);

  // 11) out = LN2(s2) -> f32
  layernorm_rows<<<SEQ, blk, 0, stream>>>(s2, ln2_g, ln2_b, out, nullptr, HID);

  (void)in_sizes; (void)n_in; (void)out_size; (void)ws_size;
}
